// SVCD_11682311045593
// MI455X (gfx1250) — matrix operand order repaired and checked by reading, not compiled, not run on hardware
//
#include <hip/hip_runtime.h>

typedef __attribute__((ext_vector_type(16))) _Float16 v16h;
typedef __attribute__((ext_vector_type(8)))  float    v8f;
typedef __attribute__((ext_vector_type(8)))  unsigned v8u;

namespace {
constexpr int   B_    = 4;
constexpr int   C_    = 4;
constexpr int   S_    = 1 << 20;      // 1024*1024 pixels per batch
constexpr int   NBINS = 64;
constexpr int   RAD   = 2;
constexpr int   NBTOT = NBINS + 2 * RAD;   // 68
constexpr int   HSZ   = C_ * NBTOT;        // 272
constexpr float EPS   = 1e-8f;

// workspace layout (32-bit slots)
constexpr int WS_KEYS = 0;               // 2*B uints: [2b]=minkey, [2b+1]=maxkey
constexpr int WS_PAR  = 8;               // 2*B floats: [8+2b]=pad_min, [9+2b]=bin_width
constexpr int WS_HIST = 16;              // B*HSZ floats
constexpr int WS_COND = 16 + B_ * HSZ;   // B*HSZ floats
}

__device__ __forceinline__ unsigned f2key(float f) {
    unsigned bits = __float_as_uint(f);
    return (bits & 0x80000000u) ? ~bits : (bits | 0x80000000u);
}
__device__ __forceinline__ float key2f(unsigned k) {
    unsigned bits = (k & 0x80000000u) ? (k & 0x7FFFFFFFu) : ~k;
    return __uint_as_float(bits);
}
__device__ __forceinline__ float bspline(float d) {
    float a = fabsf(d);
    float r = 0.f;
    if (a < 1.f)       r = (3.f * a * a * a - 6.f * a * a + 4.f) * (1.f / 6.f);
    else if (a < 2.f)  { float t = 2.f - a; r = t * t * t * (1.f / 6.f); }
    return r;
}

// ---------------------------------------------------------------- init ws
__global__ void k_init(unsigned* wsu, float* wsf) {
    int i = blockIdx.x * blockDim.x + threadIdx.x;
    if (i < 2 * B_) wsu[WS_KEYS + i] = (i & 1) ? 0u : 0xFFFFFFFFu;
    if (i >= WS_HIST && i < WS_HIST + B_ * HSZ) wsf[i] = 0.f;
}

// ---------------------------------------------------------------- masked min/max per batch
__global__ void k_minmax(const float* __restrict__ img, const int* __restrict__ msk,
                         unsigned* wsu) {
    const int b = blockIdx.y;
    __shared__ unsigned smn[256], smx[256];
    unsigned mn = 0xFFFFFFFFu, mx = 0u;
    const float* ib = img + (size_t)b * S_;
    const int*   mb = msk + (size_t)b * S_;
    for (int i = blockIdx.x * blockDim.x + threadIdx.x; i < S_; i += gridDim.x * blockDim.x) {
        if (mb[i] > 0) {
            unsigned k = f2key(ib[i]);
            mn = (k < mn) ? k : mn;
            mx = (k > mx) ? k : mx;
        }
    }
    smn[threadIdx.x] = mn; smx[threadIdx.x] = mx;
    __syncthreads();
    for (int off = 128; off > 0; off >>= 1) {
        if (threadIdx.x < (unsigned)off) {
            unsigned a = smn[threadIdx.x + off];
            if (a < smn[threadIdx.x]) smn[threadIdx.x] = a;
            unsigned c = smx[threadIdx.x + off];
            if (c > smx[threadIdx.x]) smx[threadIdx.x] = c;
        }
        __syncthreads();
    }
    if (threadIdx.x == 0) {
        atomicMin(&wsu[WS_KEYS + 2 * b],     smn[0]);
        atomicMax(&wsu[WS_KEYS + 2 * b + 1], smx[0]);
    }
}

// ---------------------------------------------------------------- bin params per batch
__global__ void k_params(const unsigned* wsu, float* wsf) {
    int b = threadIdx.x;
    if (b >= B_) return;
    float mn = key2f(wsu[WS_KEYS + 2 * b]);
    float mx = key2f(wsu[WS_KEYS + 2 * b + 1]);
    float bw = (mx - mn) / (float)NBINS;
    wsf[WS_PAR + 2 * b]     = mn - bw * (float)RAD;   // pad_min_v
    wsf[WS_PAR + 2 * b + 1] = bw;                     // bin_width
}

// ---------------------------------------------------------------- soft histogram (LDS ds_add_f32)
__global__ void __launch_bounds__(256) k_hist(const float* __restrict__ img,
                                              const float* __restrict__ wgt,
                                              const int* __restrict__ msk,
                                              float* wsf) {
    const int b = blockIdx.y;
    __shared__ float lh[HSZ];
    for (int t = threadIdx.x; t < HSZ; t += blockDim.x) lh[t] = 0.f;
    __syncthreads();

    const float pad = wsf[WS_PAR + 2 * b];
    const float bw  = wsf[WS_PAR + 2 * b + 1];
    const float inv = 1.f / fmaxf(bw, EPS);
    const float* ib = img + (size_t)b * S_;
    const int*   mb = msk + (size_t)b * S_;
    const float* wb = wgt + (size_t)b * C_ * S_;

    for (int i = blockIdx.x * blockDim.x + threadIdx.x; i < S_; i += gridDim.x * blockDim.x) {
        if (mb[i] <= 0) continue;
        float bp = (ib[i] - pad) * inv;
        float bi = fminf(fmaxf(floorf(bp), (float)RAD), (float)(NBINS - 1 + RAD));
        int   mw = (int)bi - RAD + 1;
        float w0 = wb[0 * (size_t)S_ + i];
        float w1 = wb[1 * (size_t)S_ + i];
        float w2 = wb[2 * (size_t)S_ + i];
        float w3 = wb[3 * (size_t)S_ + i];
#pragma unroll
        for (int j = 0; j < 2 * RAD; ++j) {
            int   wi = mw + j;
            float kv = bspline(bp - (float)wi);
            atomicAdd(&lh[0 * NBTOT + wi], w0 * kv);
            atomicAdd(&lh[1 * NBTOT + wi], w1 * kv);
            atomicAdd(&lh[2 * NBTOT + wi], w2 * kv);
            atomicAdd(&lh[3 * NBTOT + wi], w3 * kv);
        }
    }
    __syncthreads();
    for (int t = threadIdx.x; t < HSZ; t += blockDim.x)
        atomicAdd(&wsf[WS_HIST + b * HSZ + t], lh[t]);
}

// ---------------------------------------------------------------- normalize -> cond
__global__ void k_cond(float* wsf) {
    const int b = blockIdx.x;
    __shared__ float sh[HSZ];
    __shared__ float tot, row[C_];
    for (int t = threadIdx.x; t < HSZ; t += blockDim.x)
        sh[t] = wsf[WS_HIST + b * HSZ + t];
    __syncthreads();
    if (threadIdx.x == 0) {
        float T = 0.f;
        for (int i = 0; i < HSZ; ++i) T += sh[i];
        tot = fmaxf(T, EPS);
        for (int c = 0; c < C_; ++c) {
            float r = 0.f;
            for (int k = 0; k < NBTOT; ++k) r += sh[c * NBTOT + k];
            row[c] = r;
        }
    }
    __syncthreads();
    for (int t = threadIdx.x; t < HSZ; t += blockDim.x) {
        int   c    = t / NBTOT;
        float dens = sh[t] / tot;
        float rden = fmaxf(row[c] / tot, EPS);
        wsf[WS_COND + b * HSZ + t] = dens / rden;
    }
}

// ---------------------------------------------------------------- gather as one-hot WMMA matmul
// out[c, s] = sum_bin cond[c,bin] * onehot[bin, s]; 16 pixels / wave-iter, 3x wmma_f32_16x16x32_f16.
// A (16x32 f16): rows 0-3 = cond channels, rows 8-11 duplicate them (rest 0); loop-invariant per batch.
// B (32x16 f16): one-hot of per-pixel bin index, built word-wise (pair-index compare vs immediates).
// D (16x16 f32): lanes 0-15 store channels 0,1 (rows 0,1); lanes 16-31 store channels 2,3
// (rows 10,11) -> all 32 lanes store 2 floats, branch-free via cndmask selects.
__global__ void __launch_bounds__(256) k_gather(const float* __restrict__ img,
                                                const float* __restrict__ wsf,
                                                float* __restrict__ out) {
    __shared__ float condAll[B_ * HSZ];  // 1088 floats
    for (int t = threadIdx.x; t < B_ * HSZ; t += blockDim.x)
        condAll[t] = wsf[WS_COND + t];
    __syncthreads();

    const int lane = threadIdx.x & 31;
    const int wave = (blockIdx.x * blockDim.x + threadIdx.x) >> 5;
    const int nwv  = (gridDim.x * blockDim.x) >> 5;
    const int m    = lane & 15;            // pixel slot (cols) / A row
    const int hi   = (lane >> 4) & 1;      // lane half selects K sub-range
    const int chan = m & 7;                // A rows 0-3 and 8-11 carry channels

    for (int b = 0; b < B_; ++b) {
        const float pad = wsf[WS_PAR + 2 * b];
        const float bw  = wsf[WS_PAR + 2 * b + 1];

        // Precompute loop-invariant A tiles (cond -> f16).
        // k order used here (contiguous, B matches): lane half 'hi' holds K = {hi*16 + h}.
        v16h A[3];
#pragma unroll
        for (int T = 0; T < 3; ++T) {
#pragma unroll
            for (int h = 0; h < 16; ++h) {
                int kk  = hi * 16 + h;
                int bin = T * 32 + kk;
                float v = (chan < C_ && bin < NBTOT)
                              ? condAll[b * HSZ + chan * NBTOT + bin] : 0.f;
                A[T][h] = (_Float16)v;
            }
        }

        const float* ib = img + (size_t)b * S_;
        float*       ob = out + (size_t)b * C_ * S_;
        const int  ngrp = S_ / 16;

        for (int g = wave; g < ngrp; g += nwv) {
            const int base = g * 16;
            float iv = ib[base + m];                 // lanes 16-31 mirror lanes 0-15
            float tp = (iv - pad) / bw;              // reference uses raw bin_width here
            int  idx = (int)truncf(tp);
            idx = idx < 0 ? 0 : (idx > NBTOT - 1 ? NBTOT - 1 : idx);

            // One-hot B built per dword: bin pair jj vs compile-time immediates.
            const int      jj = (idx >> 1) - hi * 8;                       // lane-half folded in
            const unsigned hv = (idx & 1) ? 0x3C000000u : 0x3C00u;         // 1.0h in high/low half

            v8f acc = {};
#pragma unroll
            for (int T = 0; T < 3; ++T) {
                v8u Bw;
#pragma unroll
                for (int w = 0; w < 8; ++w)
                    Bw[w] = (jj == T * 16 + w) ? hv : 0u;                  // v_cmp vs inline imm
                v16h Bm = __builtin_bit_cast(v16h, Bw);
                acc = __builtin_amdgcn_wmma_f32_16x16x32_f16(
                    /*neg_a=*/false, A[T], /*neg_b=*/false, Bm,
                    /*c_mod=*/(short)0, acc, /*reuse_a=*/false, /*reuse_b=*/false);
            }

            // Branch-free epilogue: lane half picks its two channels.
            float  o0 = hi ? acc[2] : acc[0];        // row 0 (c0) or row 10 (c2)
            float  o1 = hi ? acc[3] : acc[1];        // row 1 (c1) or row 11 (c3)
            size_t p  = (size_t)(hi * 2) * S_ + (size_t)(base + m);
            ob[p]      = o0;
            ob[p + S_] = o1;
        }
    }
}

// ---------------------------------------------------------------- launch
extern "C" void kernel_launch(void* const* d_in, const int* in_sizes, int n_in,
                              void* d_out, int out_size, void* d_ws, size_t ws_size,
                              hipStream_t stream) {
    const float* img = (const float*)d_in[0];
    const float* wgt = (const float*)d_in[1];
    const int*   msk = (const int*)d_in[2];
    float*    out = (float*)d_out;
    float*    wsf = (float*)d_ws;
    unsigned* wsu = (unsigned*)d_ws;

    k_init  <<<dim3((WS_COND + 255) / 256 + 1), 256, 0, stream>>>(wsu, wsf);
    k_minmax<<<dim3(256, B_), 256, 0, stream>>>(img, msk, wsu);
    k_params<<<1, 32, 0, stream>>>(wsu, wsf);
    k_hist  <<<dim3(128, B_), 256, 0, stream>>>(img, wgt, msk, wsf);
    k_cond  <<<dim3(B_), 288, 0, stream>>>(wsf);
    k_gather<<<dim3(1024), 256, 0, stream>>>(img, wsf, out);
}
